// Word_Graph_Attention_30056181137545
// MI455X (gfx1250) — matrix operand order repaired and checked by reading, not compiled, not run on hardware
//
#include <hip/hip_runtime.h>
#include <hip/hip_bf16.h>
#include <math.h>

// Problem constants (match reference)
#define B_   32
#define E_   256
#define N_   64
#define KV   100
#define QD   768

typedef _Float16 v16h __attribute__((ext_vector_type(16)));
typedef float    v8f  __attribute__((ext_vector_type(8)));
typedef int      v4i  __attribute__((vector_size(4 * sizeof(int))));

#define GLOBAL_AS __attribute__((address_space(1)))
#define LDS_AS    __attribute__((address_space(3)))

// ---------------------------------------------------------------------------
// Kernel 1: Qp[b] = tanh(q[b] @ W_q^T + b_q);  qk[b,d] = sum_j Qp[b,j]*W_kv[j,d]
// grid = B_, block = 128
// ---------------------------------------------------------------------------
__global__ void wga_qproj_kernel(const float* __restrict__ q,
                                 const float* __restrict__ Wq,
                                 const float* __restrict__ bq,
                                 const float* __restrict__ Wkv,
                                 float* __restrict__ qkws) {
    __shared__ float sQp[KV];
    const int b = blockIdx.x;
    const int t = threadIdx.x;
    if (t < KV) {
        const float* qr = q + (size_t)b * QD;
        const float* wr = Wq + (size_t)t * QD;
        float acc = 0.f;
        for (int i = 0; i < QD; ++i) acc += qr[i] * wr[i];
        sQp[t] = tanhf(acc + bq[t]);
    }
    __syncthreads();
    if (t < KV) {
        float acc = 0.f;
        for (int j = 0; j < KV; ++j) acc += sQp[j] * Wkv[j * KV + t];
        qkws[b * KV + t] = acc;
    }
}

// ---------------------------------------------------------------------------
// Kernel 2: bandwidth kernel. One block per (b,e).
//   - stage k[b,e] (64x100 f32) and v[b,e] into LDS via async global->LDS DMA
//   - s[n] = qk . k[n] / 10 ; mask==0 -> -1e4 ; leaky_relu ; softmax ; mask==1/64 -> 0
//   - agg[c] = sum_n att[n]*v[n,c]  -> f16 workspace
// grid = B_*E_, block = 256
// ---------------------------------------------------------------------------
__global__ void wga_score_agg_kernel(const float* __restrict__ k,
                                     const float* __restrict__ v,
                                     const float* __restrict__ qkws,
                                     _Float16* __restrict__ aggws) {
    __shared__ float sK[N_ * KV];
    __shared__ float sV[N_ * KV];
    __shared__ float sQk[KV];
    __shared__ float sAtt[N_];
    __shared__ float sRed[2];

    const int be = blockIdx.x;          // b*E_ + e
    const int b  = be / E_;
    const int t  = threadIdx.x;

    // stage qk
    if (t < KV) sQk[t] = qkws[b * KV + t];

    // stage k,v tiles (6400 floats = 1600 x 16B each)
    const float4* kp4 = reinterpret_cast<const float4*>(k + (size_t)be * (N_ * KV));
    const float4* vp4 = reinterpret_cast<const float4*>(v + (size_t)be * (N_ * KV));
    float4* sK4 = reinterpret_cast<float4*>(sK);
    float4* sV4 = reinterpret_cast<float4*>(sV);
#if defined(__AMDGCN__) && __has_builtin(__builtin_amdgcn_global_load_async_to_lds_b128)
    for (int i = t; i < (N_ * KV) / 4; i += 256) {
        __builtin_amdgcn_global_load_async_to_lds_b128(
            (GLOBAL_AS v4i*)(kp4 + i), (LDS_AS v4i*)(sK4 + i), 0, 0);
        __builtin_amdgcn_global_load_async_to_lds_b128(
            (GLOBAL_AS v4i*)(vp4 + i), (LDS_AS v4i*)(sV4 + i), 0, 0);
    }
    asm volatile("s_wait_asynccnt 0" ::: "memory");
#else
    for (int i = t; i < (N_ * KV) / 4; i += 256) {
        sK4[i] = kp4[i];
        sV4[i] = vp4[i];
    }
#endif
    __syncthreads();

    // scores: 4 threads per row, 25 dims each, wave-local reduce (wave32)
    {
        const int n = t >> 2;
        const int p = t & 3;
        const float* kr = sK + n * KV + p * 25;
        const float* qr = sQk + p * 25;
        float part = 0.f;
        for (int i = 0; i < 25; ++i) part += qr[i] * kr[i];
        part += __shfl_xor(part, 1, 32);
        part += __shfl_xor(part, 2, 32);
        if (p == 0) sAtt[n] = part * 0.1f;   // / sqrt(100)
    }
    __syncthreads();

    // mask(att==0 -> -1e4) then leaky_relu(0.01)
    if (t < N_) {
        float x = sAtt[t];
        if (x == 0.0f) x = -10000.0f;
        x = (x > 0.0f) ? x : 0.01f * x;
        sAtt[t] = x;
    }
    __syncthreads();
    if (t == 0) {
        float mx = sAtt[0];
        for (int n = 1; n < N_; ++n) mx = fmaxf(mx, sAtt[n]);
        sRed[0] = mx;
    }
    __syncthreads();
    if (t < N_) sAtt[t] = expf(sAtt[t] - sRed[0]);
    __syncthreads();
    if (t == 0) {
        float s = 0.f;
        for (int n = 0; n < N_; ++n) s += sAtt[n];
        sRed[1] = s;
    }
    __syncthreads();
    if (t < N_) {
        float p = sAtt[t] / sRed[1];
        if (p == (1.0f / (float)N_)) p = 0.0f;   // reference mask
        sAtt[t] = p;
    }
    __syncthreads();

    // agg[c] = sum_n att[n] * v[n,c]
    if (t < KV) {
        float acc = 0.f;
        for (int n = 0; n < N_; ++n) acc += sAtt[n] * sV[n * KV + t];
        aggws[(size_t)be * KV + t] = (_Float16)acc;
    }
}

// ---------------------------------------------------------------------------
// Kernel 3: out[r,j] = sum_c agg[r,c] * W_v[j,c]   (r in [0,8192), j in [0,100))
// WMMA f32_16x16x32_f16, K padded to 128, N padded to 112 (7 tiles).
// LDS holds A and B in *fragment-ready* per-lane order so each lane loads its
// whole 16-half fragment as one contiguous 32B read (ds_load_b128 pairs).
// grid = 8192/16 = 512 blocks of 256 threads (8 waves; waves 0..6 own N-tiles).
// ---------------------------------------------------------------------------
__global__ void wga_outproj_wmma_kernel(const _Float16* __restrict__ aggws,
                                        const float* __restrict__ Wv,
                                        float* __restrict__ out) {
    // A fragments: [kc(4)][lane(32)][e(16)]   = 2048 halfs (4 KB)
    // B fragments: [nt(7)][kc(4)][lane(32)][e(16)] = 14336 halfs (28 KB)
    __shared__ __align__(32) _Float16 sAf[4 * 32 * 16];
    __shared__ __align__(32) _Float16 sWf[7 * 4 * 32 * 16];

    const int t     = threadIdx.x;
    const int mbase = blockIdx.x * 16;       // first agg row of this block

    // ---- stage W_v^T into B-fragment order ----
    // B element (K=kc*32 + ((lane>>4)<<3) + (e&7) + ((e&8)<<1), Ncol = nt*16 + (lane&15))
    for (int idx = t; idx < 7 * 4 * 32 * 16; idx += 256) {
        const int e    = idx & 15;
        const int lane = (idx >> 4) & 31;
        const int kc   = (idx >> 9) & 3;
        const int nt   = idx >> 11;
        const int K    = kc * 32 + ((lane >> 4) << 3) + (e & 7) + ((e & 8) << 1);
        const int j    = nt * 16 + (lane & 15);
        float val = (K < KV && j < KV) ? Wv[j * KV + K] : 0.0f;
        sWf[idx] = (_Float16)val;
    }
    // ---- stage 16 agg rows into A-fragment order ----
    // A element (row r, K kk): kc=kk>>5; kk32=kk&31;
    //   lane = ((kk32>>3)&1)*16 + r;  e = (kk32&7) + ((kk32&16)>>1)
    for (int idx = t; idx < 16 * 128; idx += 256) {
        const int r    = idx >> 7;
        const int kk   = idx & 127;
        _Float16 val = (_Float16)0.0f;
        if (kk < KV) val = aggws[(size_t)(mbase + r) * KV + kk];
        const int kc   = kk >> 5;
        const int kk32 = kk & 31;
        const int lane = (((kk32 >> 3) & 1) << 4) + r;
        const int e    = (kk32 & 7) + ((kk32 & 16) >> 1);
        sAf[(kc * 32 + lane) * 16 + e] = val;
    }
    __syncthreads();

    const int wave = t >> 5;                 // 0..7
    if (wave < 7) {
        const int nt   = wave;               // N tile: cols [nt*16, nt*16+16)
        const int lane = t & 31;
        const int hi   = lane >> 4;
        const int l16  = lane & 15;

        const v16h* Afp = reinterpret_cast<const v16h*>(sAf);
        const v16h* Bfp = reinterpret_cast<const v16h*>(sWf);

        v8f acc = {};
        #pragma unroll
        for (int kc = 0; kc < 4; ++kc) {     // K chunks of 32
            v16h afrag = Afp[kc * 32 + lane];
            v16h bfrag = Bfp[(nt * 4 + kc) * 32 + lane];
            acc = __builtin_amdgcn_wmma_f32_16x16x32_f16(
                    /*neg_a=*/false, afrag, /*neg_b=*/false, bfrag,
                    /*c_mod=*/(short)0, acc, /*reuse_a=*/false, /*reuse_b=*/false);
        }

        // C layout: VGPR i -> row (hi?8:0)+i, col = nt*16 + l16
        const int col = nt * 16 + l16;
        if (col < KV) {
            #pragma unroll
            for (int i = 0; i < 8; ++i) {
                const int row = mbase + (hi ? 8 : 0) + i;
                out[(size_t)row * KV + col] = acc[i];
            }
        }
    }
}

// ---------------------------------------------------------------------------
extern "C" void kernel_launch(void* const* d_in, const int* in_sizes, int n_in,
                              void* d_out, int out_size, void* d_ws, size_t ws_size,
                              hipStream_t stream) {
    // inputs: input_ent(int, unused), q, k, v, W_q, b_q, W_kv, W_v
    const float* q   = (const float*)d_in[1];
    const float* k   = (const float*)d_in[2];
    const float* v   = (const float*)d_in[3];
    const float* Wq  = (const float*)d_in[4];
    const float* bq  = (const float*)d_in[5];
    const float* Wkv = (const float*)d_in[6];
    const float* Wv  = (const float*)d_in[7];
    float* out = (float*)d_out;

    // workspace: qk [32x100] f32, then agg [8192x100] f16
    float*    qkws  = (float*)d_ws;
    _Float16* aggws = (_Float16*)((char*)d_ws + B_ * KV * sizeof(float));

    wga_qproj_kernel<<<B_, 128, 0, stream>>>(q, Wq, bq, Wkv, qkws);
    wga_score_agg_kernel<<<B_ * E_, 256, 0, stream>>>(k, v, qkws, aggws);
    wga_outproj_wmma_kernel<<<(B_ * E_) / 16, 256, 0, stream>>>(aggws, Wv, out);
}
